// Block_65841848648273
// MI455X (gfx1250) — compile-verified, matrix operand order checked by reading, not run on hardware
//
#include <hip/hip_runtime.h>

// ---------------- problem constants ----------------
#define T_    4
#define B_    16
#define N_    256
#define C_    512
#define H_    8
#define DH_   64
#define HID_  2048
#define EPS_  1e-5f
#define SCALE_ 0.125f
#define R_    (T_*B_*N_)        // 16384 rows flattened
#define SSPLIT 32               // row splits for BN stats

typedef __attribute__((ext_vector_type(16))) __bf16 v16bf;
typedef __attribute__((ext_vector_type(8)))  float  v8f;
typedef __attribute__((ext_vector_type(4)))  unsigned int v4u;
typedef __attribute__((ext_vector_type(8)))  int v8i;
typedef __attribute__((ext_vector_type(4)))  int v4i;

#if defined(__has_builtin)
#  if __has_builtin(__builtin_amdgcn_tensor_load_to_lds)
#    define HAVE_TDM 1
#  endif
#endif

// D = A(16x32 bf16) * B(32x16 bf16) + C(16x16 f32)
static __device__ inline v8f wmma_bf16(v16bf a, v16bf b, v8f c) {
  return __builtin_amdgcn_wmma_f32_16x16x32_bf16(
      /*neg_a=*/false, a, /*neg_b=*/false, b,
      /*c_mod=*/(short)0, c, /*reuse_a=*/false, /*reuse_b=*/false);
}

// ISA 16-bit A-matrix 16x32 layout: lane half h = lane>>4, row r = lane&15.
// element j (0..15): K = 8*h + 16*(j>>3) + (j&7); value = base[r*ld + k]
static __device__ inline v16bf frag_rm(const __bf16* base, int ld, int lane) {
  int r = lane & 15, h = (lane >> 4) << 3;
  v16bf f;
#pragma unroll
  for (int j = 0; j < 16; ++j) {
    int k = h + ((j & 8) << 1) + (j & 7);
    f[j] = base[r * ld + k];
  }
  return f;
}
// same lane/K pattern, element(r, K) = base[k*ld + r]  (column-major read)
static __device__ inline v16bf frag_cm(const __bf16* base, int ld, int lane) {
  int r = lane & 15, h = (lane >> 4) << 3;
  v16bf f;
#pragma unroll
  for (int j = 0; j < 16; ++j) {
    int k = h + ((j & 8) << 1) + (j & 7);
    f[j] = base[k * ld + r];
  }
  return f;
}

// ---------------- TDM 2-D tile load: global -> LDS ----------------
// D# per CDNA5 ISA. 2-D tile, bf16 (data_size code 1),
// groups 2/3 zero (tile_dim2/3 unused). lds_off is the byte offset in LDS
// (attention kernel has no static LDS, so dynamic LDS starts at 0).
static __device__ inline void tdm_load_2d(unsigned lds_off, const void* gaddr,
                                          unsigned tile_x, unsigned tile_y,
                                          unsigned stride_elems) {
#if defined(HAVE_TDM)
  unsigned long long ga = (unsigned long long)(uintptr_t)gaddr;
  v4u g0;
  g0[0] = 1u;                                          // count=1 (valid), user mode
  g0[1] = lds_off;                                     // lds_addr (bytes)
  g0[2] = (unsigned)(ga & 0xffffffffu);                // global_addr[31:0]
  g0[3] = (unsigned)((ga >> 32) & 0x01ffffffu) | (2u << 30);  // addr[56:32], type=2
  unsigned td0 = 1u << 20, td1 = 1u << 20;             // huge tensor dims: no OOB clip
  v8i g1;
  g1[0] = (int)(1u << 16);                             // wg_mask=0, data_size=1 (2B)
  g1[1] = (int)((td0 & 0xffffu) << 16);                // tensor_dim0[15:0] @ bits 63:48
  g1[2] = (int)((td0 >> 16) | ((td1 & 0xffffu) << 16));// tdim0[31:16] | tdim1[15:0]
  g1[3] = (int)((td1 >> 16) | (tile_x << 16));         // tdim1[31:16] | tile_dim0
  g1[4] = (int)tile_y;                                 // tile_dim1 (tile_dim2=0)
  g1[5] = (int)stride_elems;                           // tensor_dim0_stride[31:0]
  g1[6] = 0;                                           // stride0[47:32]=0, stride1 lo=0
  g1[7] = 0;
  v4i z4 = (v4i)0;
#if __clang_major__ >= 23
  v8i z8 = (v8i)0;
  __builtin_amdgcn_tensor_load_to_lds(g0, g1, z4, z4, z8, 0);
#else
  __builtin_amdgcn_tensor_load_to_lds(g0, g1, z4, z4, 0);
#endif
#else
  (void)lds_off; (void)gaddr; (void)tile_x; (void)tile_y; (void)stride_elems;
#endif
}

// ---------------- elementwise: fp32 -> bf16 ----------------
__global__ void k_f32_to_bf16(const float* __restrict__ in, __bf16* __restrict__ out, long n) {
  long i = (long)blockIdx.x * blockDim.x + threadIdx.x;
  if (i < n) out[i] = (__bf16)in[i];
}

// ---------------- GEMM: Y[M,Nout] = A(bf16)[M,K] @ W[Nout,K]^T + bias ----------------
// 256 threads = 8 waves; 128x128 block tile; wave (4x2 grid) owns 32x64;
// double-buffered LDS, one barrier per K-step, 8 WMMAs per wave per step.
__global__ void k_gemm(const __bf16* __restrict__ A, const float* __restrict__ W,
                       const float* __restrict__ bias, float* __restrict__ Y,
                       int K, int Nout) {
  __shared__ __bf16 As[2 * 128 * 32];
  __shared__ __bf16 Ws[2 * 128 * 32];
  int tid  = threadIdx.x;
  int lane = tid & 31, wave = tid >> 5;
  int wr = (wave >> 1) * 32;          // wave row offset (0,32,64,96)
  int wc = (wave & 1) * 64;           // wave col offset (0,64)
  long m0 = (long)blockIdx.x * 128;
  long n0 = (long)blockIdx.y * 128;
  int lr = tid >> 1;                  // 0..127 tile row for loads
  int lc = (tid & 1) * 16;            // 0 or 16

  auto loadTiles = [&](int buf, int kk) {
    __bf16* Ad = As + buf * 4096;
    __bf16* Wd = Ws + buf * 4096;
    const __bf16* ap = A + (m0 + lr) * (long)K + kk + lc;
    *reinterpret_cast<uint4*>(&Ad[lr * 32 + lc])     = *reinterpret_cast<const uint4*>(ap);
    *reinterpret_cast<uint4*>(&Ad[lr * 32 + lc + 8]) = *reinterpret_cast<const uint4*>(ap + 8);
    const float* wp = W + (n0 + lr) * (long)K + kk + lc;
#pragma unroll
    for (int j = 0; j < 16; ++j) Wd[lr * 32 + lc + j] = (__bf16)wp[j];
  };

  v8f acc0[4] = {}; v8f acc1[4] = {};
  int nsteps = K >> 5;
  loadTiles(0, 0);
  for (int s = 0; s < nsteps; ++s) {
    __syncthreads();
    if (s + 1 < nsteps) loadTiles((s + 1) & 1, (s + 1) << 5);
    const __bf16* Ab = As + (s & 1) * 4096;
    const __bf16* Wb = Ws + (s & 1) * 4096;
    v16bf a0 = frag_rm(Ab + (wr +  0) * 32, 32, lane);
    v16bf a1 = frag_rm(Ab + (wr + 16) * 32, 32, lane);
#pragma unroll
    for (int cb = 0; cb < 4; ++cb) {
      v16bf b = frag_rm(Wb + (wc + cb * 16) * 32, 32, lane);
      acc0[cb] = wmma_bf16(a0, b, acc0[cb]);
      acc1[cb] = wmma_bf16(a1, b, acc1[cb]);
    }
  }
  int half = lane >> 4, col = lane & 15;
#pragma unroll
  for (int cb = 0; cb < 4; ++cb) {
    long n = n0 + wc + cb * 16 + col;
    float bv = bias[n];
#pragma unroll
    for (int i = 0; i < 8; ++i) {
      long m = m0 + wr + i + 8 * half;
      Y[m * Nout + n]        = acc0[cb][i] + bv;
      Y[(m + 16) * Nout + n] = acc1[cb][i] + bv;
    }
  }
}

// ---------------- BN stats, coalesced two-stage reduction ----------------
// stage 1: grid (Nout/64, SSPLIT); block 256 = 4 row-threads x 64 channels
__global__ void k_colstats_part(const float* __restrict__ Y, float* __restrict__ psum,
                                float* __restrict__ psq, int M, int Nout) {
  int n  = blockIdx.x * 64 + (threadIdx.x & 63);
  int rr = threadIdx.x >> 6;                 // 0..3
  int rows = M / SSPLIT;
  int r0 = blockIdx.y * rows;
  float s = 0.f, q = 0.f;
  for (int r = r0 + rr; r < r0 + rows; r += 4) {
    float y = Y[(long)r * Nout + n];
    s += y; q += y * y;
  }
  __shared__ float ls[256], lq[256];
  ls[threadIdx.x] = s; lq[threadIdx.x] = q;
  __syncthreads();
  if (rr == 0) {
    int t = threadIdx.x;
    s = ls[t] + ls[t + 64] + ls[t + 128] + ls[t + 192];
    q = lq[t] + lq[t + 64] + lq[t + 128] + lq[t + 192];
    psum[(long)blockIdx.y * Nout + n] = s;
    psq [(long)blockIdx.y * Nout + n] = q;
  }
}
// stage 2: scale = g*rsqrt(var+eps); shift = beta - mean*scale
__global__ void k_colstats_fin(const float* __restrict__ psum, const float* __restrict__ psq,
                               const float* __restrict__ g, const float* __restrict__ beta,
                               float* __restrict__ scale, float* __restrict__ shift,
                               int M, int Nout) {
  int n = blockIdx.x * blockDim.x + threadIdx.x;
  if (n >= Nout) return;
  float s = 0.f, q = 0.f;
#pragma unroll
  for (int j = 0; j < SSPLIT; ++j) { s += psum[(long)j * Nout + n]; q += psq[(long)j * Nout + n]; }
  float mean = s / (float)M;
  float var  = q / (float)M - mean * mean;
  float sc   = g[n] * rsqrtf(var + EPS_);
  scale[n] = sc;
  shift[n] = beta[n] - mean * sc;
}

// ---------------- BN + LIF scan over T, spikes -> bf16 ----------------
__global__ void k_bn_lif_bf16(const float* __restrict__ Y, const float* __restrict__ scale,
                              const float* __restrict__ shift, __bf16* __restrict__ out,
                              long per_t, int Cn) {
  long i = (long)blockIdx.x * blockDim.x + threadIdx.x;
  if (i >= per_t) return;
  int c = (int)(i % Cn);
  float sc = scale[c], sh = shift[c];
  float v = 0.f;
#pragma unroll
  for (int t = 0; t < T_; ++t) {
    float y = Y[(long)t * per_t + i] * sc + sh;
    v = 0.5f * (v + y);                 // v += (x - v)/tau, tau = 2
    float s = (v >= 1.f) ? 1.f : 0.f;   // hard threshold
    out[(long)t * per_t + i] = (__bf16)s;
    v *= (1.f - s);                     // hard reset
  }
}

// ---------------- BN only -> bf16 (v path) ----------------
__global__ void k_bn_bf16(const float* __restrict__ Y, const float* __restrict__ scale,
                          const float* __restrict__ shift, __bf16* __restrict__ out,
                          long n, int Cn) {
  long i = (long)blockIdx.x * blockDim.x + threadIdx.x;
  if (i >= n) return;
  int c = (int)(i % Cn);
  out[i] = (__bf16)(Y[i] * scale[c] + shift[c]);
}

// ---------------- BN + residual: x1 = x + BN(Y); fp32 to d_out + bf16 copy ----------------
__global__ void k_bn_resid(const float* __restrict__ Y, const float* __restrict__ scale,
                           const float* __restrict__ shift, const float* __restrict__ x,
                           float* __restrict__ out, __bf16* __restrict__ outb,
                           long n, int Cn) {
  long i = (long)blockIdx.x * blockDim.x + threadIdx.x;
  if (i >= n) return;
  int c = (int)(i % Cn);
  float r = x[i] + Y[i] * scale[c] + shift[c];
  out[i]  = r;
  outb[i] = (__bf16)r;
}

// ---------------- BN + LIF, spike added into d_out (final x1 + m) ----------------
__global__ void k_bn_lif_add(const float* __restrict__ Y, const float* __restrict__ scale,
                             const float* __restrict__ shift, float* __restrict__ out,
                             long per_t, int Cn) {
  long i = (long)blockIdx.x * blockDim.x + threadIdx.x;
  if (i >= per_t) return;
  int c = (int)(i % Cn);
  float sc = scale[c], sh = shift[c];
  float v = 0.f;
#pragma unroll
  for (int t = 0; t < T_; ++t) {
    float y = Y[(long)t * per_t + i] * sc + sh;
    v = 0.5f * (v + y);
    float s = (v >= 1.f) ? 1.f : 0.f;
    out[(long)t * per_t + i] += s;
    v *= (1.f - s);
  }
}

// ---------------- Attention: one workgroup per (t,b,head) ----------------
// o = (q k^T * 0.125) v for a 256x64 head. q,k,v staged in 96KB dynamic LDS
// via TDM (wave 0 issues tensor_load_to_lds, waits TENSORcnt) or vector copies.
__global__ void k_attn(const __bf16* __restrict__ Q, const __bf16* __restrict__ Kb,
                       const __bf16* __restrict__ V, __bf16* __restrict__ O) {
  extern __shared__ __bf16 sm[];
  __bf16* qs = sm;                  // 256*64  (LDS byte off 0)
  __bf16* ks = sm + 16384;          // byte off 32768
  __bf16* vs = sm + 32768;          // byte off 65536
  __bf16* st = sm + 49152;          // 8 waves * 32*32 stage
  int tid = threadIdx.x, lane = tid & 31, wave = tid >> 5;
  int h  = blockIdx.x % H_;
  int tb = blockIdx.x / H_;         // t*B + b
  long base = ((long)tb * N_) * C_ + (long)h * DH_;
#if defined(HAVE_TDM)
  if (wave == 0) {
    tdm_load_2d(0u,     Q  + base, DH_, N_, C_);
    tdm_load_2d(32768u, Kb + base, DH_, N_, C_);
    tdm_load_2d(65536u, V  + base, DH_, N_, C_);
    __builtin_amdgcn_s_wait_tensorcnt(0);
  }
#else
  {
    const uint4* gq = reinterpret_cast<const uint4*>(Q  + base + (long)tid * C_);
    const uint4* gk = reinterpret_cast<const uint4*>(Kb + base + (long)tid * C_);
    const uint4* gv = reinterpret_cast<const uint4*>(V  + base + (long)tid * C_);
    uint4* lq = reinterpret_cast<uint4*>(qs + tid * 64);
    uint4* lk = reinterpret_cast<uint4*>(ks + tid * 64);
    uint4* lv = reinterpret_cast<uint4*>(vs + tid * 64);
#pragma unroll
    for (int j = 0; j < 8; ++j) { lq[j] = gq[j]; lk[j] = gk[j]; lv[j] = gv[j]; }
  }
#endif
  __syncthreads();
  int m0 = wave * 32;               // this wave's 32 output rows
  v8f acc[2][4] = {};               // [rowHalf][dTile] 32x64 f32 accumulator
  __bf16* stw = st + wave * 1024;   // wave-private stage (no barrier needed)
  for (int jj = 0; jj < 8; ++jj) {  // key blocks of 32
#pragma unroll
    for (int rh = 0; rh < 2; ++rh) {
#pragma unroll
      for (int ch = 0; ch < 2; ++ch) {
        v8f s = {};
#pragma unroll
        for (int kd = 0; kd < 2; ++kd) {   // Dh = 64 = 2 x K32
          v16bf a = frag_rm(qs + (m0 + rh * 16) * 64 + kd * 32, 64, lane);
          v16bf b = frag_rm(ks + (jj * 32 + ch * 16) * 64 + kd * 32, 64, lane);
          s = wmma_bf16(a, b, s);
        }
        int half = lane >> 4, col = lane & 15;
#pragma unroll
        for (int i = 0; i < 8; ++i)
          stw[(rh * 16 + i + 8 * half) * 32 + ch * 16 + col] = (__bf16)(s[i] * SCALE_);
      }
    }
#pragma unroll
    for (int rh = 0; rh < 2; ++rh) {
      v16bf a = frag_rm(stw + rh * 16 * 32, 32, lane);   // attn block as A-frag
#pragma unroll
      for (int dt = 0; dt < 4; ++dt) {
        v16bf b = frag_cm(vs + jj * 32 * 64 + dt * 16, 64, lane);  // v column tile
        acc[rh][dt] = wmma_bf16(a, b, acc[rh][dt]);
      }
    }
  }
  int half = lane >> 4, col = lane & 15;
#pragma unroll
  for (int rh = 0; rh < 2; ++rh)
#pragma unroll
    for (int dt = 0; dt < 4; ++dt)
#pragma unroll
      for (int i = 0; i < 8; ++i) {
        int row = m0 + rh * 16 + i + 8 * half;
        O[base + (long)row * C_ + dt * 16 + col] = (__bf16)acc[rh][dt][i];
      }
}

// ---------------- host orchestration ----------------
extern "C" void kernel_launch(void* const* d_in, const int* in_sizes, int n_in,
                              void* d_out, int out_size, void* d_ws, size_t ws_size,
                              hipStream_t stream) {
  const float* x  = (const float*)d_in[0];
  // d_in[1] = s_x (unused in forward, matches reference)
  const float* qw = (const float*)d_in[2];  const float* qb = (const float*)d_in[3];
  const float* qg = (const float*)d_in[4];  const float* qbe = (const float*)d_in[5];
  const float* kw = (const float*)d_in[6];  const float* kb = (const float*)d_in[7];
  const float* kg = (const float*)d_in[8];  const float* kbe = (const float*)d_in[9];
  const float* vw = (const float*)d_in[10]; const float* vb = (const float*)d_in[11];
  const float* vg = (const float*)d_in[12]; const float* vbe = (const float*)d_in[13];
  const float* pw = (const float*)d_in[14]; const float* pb = (const float*)d_in[15];
  const float* pg = (const float*)d_in[16]; const float* pbe = (const float*)d_in[17];
  const float* f1w = (const float*)d_in[18]; const float* f1b = (const float*)d_in[19];
  const float* f1g = (const float*)d_in[20]; const float* f1be = (const float*)d_in[21];
  const float* f2w = (const float*)d_in[22]; const float* f2b = (const float*)d_in[23];
  const float* f2g = (const float*)d_in[24]; const float* f2be = (const float*)d_in[25];
  float* out = (float*)d_out;

  char* ws = (char*)d_ws;
  size_t off = 0;
  auto alloc = [&](size_t bytes) -> char* {
    char* p = ws + off;
    off += (bytes + 255) & ~(size_t)255;
    return p;
  };
  float*  Y    = (float*) alloc((size_t)R_ * HID_ * 4);  // GEMM output scratch
  __bf16* xb   = (__bf16*)alloc((size_t)R_ * C_ * 2);
  __bf16* qs   = (__bf16*)alloc((size_t)R_ * C_ * 2);
  __bf16* ks   = (__bf16*)alloc((size_t)R_ * C_ * 2);
  __bf16* vs   = (__bf16*)alloc((size_t)R_ * C_ * 2);
  __bf16* ob   = (__bf16*)alloc((size_t)R_ * C_ * 2);
  __bf16* x1b  = (__bf16*)alloc((size_t)R_ * C_ * 2);
  __bf16* hb   = (__bf16*)alloc((size_t)R_ * HID_ * 2);
  float* psum  = (float*)alloc((size_t)SSPLIT * HID_ * 4);
  float* psq   = (float*)alloc((size_t)SSPLIT * HID_ * 4);
  float* scale = (float*)alloc(HID_ * 4);
  float* shift = (float*)alloc(HID_ * 4);
  (void)ws_size; (void)in_sizes; (void)n_in; (void)out_size;

  const long RC   = (long)R_ * C_;
  const long perC = (long)B_ * N_ * C_;    // per-timestep elems, C channels
  const long perH = (long)B_ * N_ * HID_;  // per-timestep elems, HID channels
  dim3 blk(256);
  dim3 gC(R_ / 128, C_ / 128);    // 128 x 4
  dim3 gH(R_ / 128, HID_ / 128);  // 128 x 16
  dim3 gsC(C_ / 64, SSPLIT), gsH(HID_ / 64, SSPLIT);

  auto stats = [&](const float* g, const float* be, int Nout, dim3 gs) {
    k_colstats_part<<<gs, blk, 0, stream>>>(Y, psum, psq, R_, Nout);
    k_colstats_fin<<<(Nout + 255) / 256, blk, 0, stream>>>(psum, psq, g, be, scale, shift, R_, Nout);
  };

  k_f32_to_bf16<<<(RC + 255) / 256, blk, 0, stream>>>(x, xb, RC);

  // --- q ---
  k_gemm<<<gC, blk, 0, stream>>>(xb, qw, qb, Y, C_, C_);
  stats(qg, qbe, C_, gsC);
  k_bn_lif_bf16<<<(perC + 255) / 256, blk, 0, stream>>>(Y, scale, shift, qs, perC, C_);
  // --- k ---
  k_gemm<<<gC, blk, 0, stream>>>(xb, kw, kb, Y, C_, C_);
  stats(kg, kbe, C_, gsC);
  k_bn_lif_bf16<<<(perC + 255) / 256, blk, 0, stream>>>(Y, scale, shift, ks, perC, C_);
  // --- v (no LIF) ---
  k_gemm<<<gC, blk, 0, stream>>>(xb, vw, vb, Y, C_, C_);
  stats(vg, vbe, C_, gsC);
  k_bn_bf16<<<(RC + 255) / 256, blk, 0, stream>>>(Y, scale, shift, vs, RC, C_);
  // --- attention (512 head-batches, 114688 B dynamic LDS, fits 320KB WGP LDS) ---
  size_t smem = (size_t)(3 * 16384 + 8 * 1024) * sizeof(__bf16);
  k_attn<<<T_ * B_ * H_, blk, smem, stream>>>(qs, ks, vs, ob);
  // --- projection + residual ---
  k_gemm<<<gC, blk, 0, stream>>>(ob, pw, pb, Y, C_, C_);
  stats(pg, pbe, C_, gsC);
  k_bn_resid<<<(RC + 255) / 256, blk, 0, stream>>>(Y, scale, shift, x, out, x1b, RC, C_);
  // --- MLP ---
  k_gemm<<<gH, blk, 0, stream>>>(x1b, f1w, f1b, Y, C_, HID_);
  stats(f1g, f1be, HID_, gsH);
  k_bn_lif_bf16<<<(perH + 255) / 256, blk, 0, stream>>>(Y, scale, shift, hb, perH, HID_);
  k_gemm<<<gC, blk, 0, stream>>>(hb, f2w, f2b, Y, HID_, C_);
  stats(f2g, f2be, C_, gsC);
  k_bn_lif_add<<<(perC + 255) / 256, blk, 0, stream>>>(Y, scale, shift, out, perC, C_);
}
